// MultiHeadAttention_31009663877885
// MI455X (gfx1250) — compile-verified, matrix operand order checked by reading, not run on hardware
//
#include <hip/hip_runtime.h>
#include <hip/hip_bf16.h>

// ---------------------------------------------------------------------------
// MI455X (gfx1250) fused multi-head attention, bf16 WMMA path.
//   B=2, S=2048, D=1024, H=16, dk=64
// Pipeline:
//   1) fp32->bf16 pre-pass for activations + weights (one pass, BW-bound)
//   2) QKV projections as bf16 WMMA GEMMs (V stored head-transposed)
//   3) flash attention (scores^T trick: softmax lane-local except xor-16 max/sum shuffles)
//   4) output projection -> fp32
// Workspace (bf16 elements): Xq(4M,->AO) Xk(4M) Xv(4M) W1..4(1M ea) Qb(4M)
// Kb(4M) Vt(4M)  => 28M elems = 56 MB, all L2-resident (192 MB).
// ---------------------------------------------------------------------------

typedef __attribute__((ext_vector_type(16))) __bf16 v16bf;
typedef __attribute__((ext_vector_type(8)))  __bf16 v8bf;
typedef __attribute__((ext_vector_type(8)))  float  v8f;

__device__ __forceinline__ v8f wmma_bf16(v16bf a, v16bf b, v8f c) {
  return __builtin_amdgcn_wmma_f32_16x16x32_bf16(false, a, false, b,
                                                 (short)0, c, false, false);
}

__device__ __forceinline__ unsigned pack2_bf16(float x, float y) {
  union { __bf16 h; unsigned short u; } a, b;
  a.h = (__bf16)x; b.h = (__bf16)y;
  return (unsigned)a.u | ((unsigned)b.u << 16);
}

// 16-bit A/B fragment from bf16 row-major source (two 16B loads):
// lane L -> row (L&15); element j -> k = k0 + (j/8)*16 + (L>>4)*8 + (j%8).
__device__ __forceinline__ v16bf frag_from_bf16(const __bf16* __restrict__ base,
                                                int row, int ld, int k0, int lane) {
  const int r  = row + (lane & 15);
  const int kb = k0 + ((lane >> 4) << 3);
  const __bf16* p = base + (size_t)r * (size_t)ld + kb;
  v8bf lo = *reinterpret_cast<const v8bf*>(p);
  v8bf hi = *reinterpret_cast<const v8bf*>(p + 16);
  v16bf f;
#pragma unroll
  for (int j = 0; j < 8; ++j) { f[j] = lo[j]; f[8 + j] = hi[j]; }
  return f;
}

// ---------------------------------------------------------------------------
// Kernel 0: elementwise fp32 -> bf16 (8 elements/thread, vectorized)
// ---------------------------------------------------------------------------
__global__ __launch_bounds__(256) void cvt_f32_bf16_kernel(
    const float* __restrict__ src, __bf16* __restrict__ dst, int n8) {
  const int i = blockIdx.x * blockDim.x + threadIdx.x;
  if (i >= n8) return;
  const float4* s = reinterpret_cast<const float4*>(src) + (size_t)i * 2;
  const float4 a = s[0], b = s[1];
  uint4 v;
  v.x = pack2_bf16(a.x, a.y);
  v.y = pack2_bf16(a.z, a.w);
  v.z = pack2_bf16(b.x, b.y);
  v.w = pack2_bf16(b.z, b.w);
  reinterpret_cast<uint4*>(dst)[i] = v;
}

// ---------------------------------------------------------------------------
// Projection GEMM:  Out = X @ W^T + bias   (bf16 in, bf16 out)
// Wave computes a 32x64 tile (8 WMMAs per k-step, 6 fragment loads).
// transposeV=0: Out row-major [4096,1024]
// transposeV=1: Out[(b*1024 + n)*2048 + s]  (V stored head-transposed)
// ---------------------------------------------------------------------------
__device__ __forceinline__ void store_proj(v8f c, const float* __restrict__ bias,
                                           __bf16* __restrict__ Out,
                                           int tm, int tn, int lane, int transposeV) {
  const int n  = tn + (lane & 15);
  const float bn = bias[n];
  const int m0 = tm + ((lane >> 4) << 3);
  if (!transposeV) {
#pragma unroll
    for (int j = 0; j < 8; ++j)
      Out[(size_t)(m0 + j) * 1024 + n] = (__bf16)(c[j] + bn);
  } else {
    const int b = m0 >> 11;      // batch
    const int s = m0 & 2047;     // sequence position
    uint4 v;
    v.x = pack2_bf16(c[0] + bn, c[1] + bn);
    v.y = pack2_bf16(c[2] + bn, c[3] + bn);
    v.z = pack2_bf16(c[4] + bn, c[5] + bn);
    v.w = pack2_bf16(c[6] + bn, c[7] + bn);
    *reinterpret_cast<uint4*>(Out + (size_t)(b * 1024 + n) * 2048 + s) = v;
  }
}

__global__ __launch_bounds__(256) void proj_qkv_kernel(
    const __bf16* __restrict__ X, const __bf16* __restrict__ W,
    const float* __restrict__ bias, __bf16* __restrict__ Out, int transposeV) {
  const int lane = threadIdx.x & 31;
  const int wave = blockIdx.x * (blockDim.x >> 5) + (threadIdx.x >> 5);
  const int tm = (wave >> 4) * 32;   // 128 row tiles
  const int tn = (wave & 15) * 64;   // 16 col tiles
  v8f c[2][4] = {};
  for (int k0 = 0; k0 < 1024; k0 += 32) {
    const v16bf a0 = frag_from_bf16(X, tm,      1024, k0, lane);
    const v16bf a1 = frag_from_bf16(X, tm + 16, 1024, k0, lane);
#pragma unroll
    for (int j = 0; j < 4; ++j) {
      const v16bf bj = frag_from_bf16(W, tn + j * 16, 1024, k0, lane);
      c[0][j] = wmma_bf16(a0, bj, c[0][j]);
      c[1][j] = wmma_bf16(a1, bj, c[1][j]);
    }
  }
#pragma unroll
  for (int i = 0; i < 2; ++i)
#pragma unroll
    for (int j = 0; j < 4; ++j)
      store_proj(c[i][j], bias, Out, tm + i * 16, tn + j * 16, lane, transposeV);
}

// ---------------------------------------------------------------------------
// Flash attention. One wave = 16 queries of one (b,h).
// scores^T = K_blk(16x64) x Q^T(64x16): each lane owns one query's scores.
// Key blocks in pairs of 16 -> the two score D-fragments map same-lane onto
// the 32x16 bf16 P^T B-fragment for out^T = Vt x P^T. No cross-lane moves.
// ---------------------------------------------------------------------------
__global__ __launch_bounds__(256) void flash_attn_kernel(
    const __bf16* __restrict__ Qb, const __bf16* __restrict__ Kb,
    const __bf16* __restrict__ Vt, __bf16* __restrict__ AO) {
  const int lane = threadIdx.x & 31;
  const int wave = blockIdx.x * (blockDim.x >> 5) + (threadIdx.x >> 5);
  const int b  = wave >> 11;          // / (16 heads * 128 qtiles)
  const int h  = (wave >> 7) & 15;
  const int q0 = (wave & 127) * 16;
  const int rowQ = b * 2048 + q0;
  const int colH = h * 64;

  const v16bf qf0 = frag_from_bf16(Qb, rowQ, 1024, colH,      lane);
  const v16bf qf1 = frag_from_bf16(Qb, rowQ, 1024, colH + 32, lane);

  v8f o0 = {}, o1 = {}, o2 = {}, o3 = {};
  float m_run = -INFINITY, l_run = 0.0f;
  const float scale = 0.125f;         // 1/sqrt(64)
  const int vrow = b * 1024 + colH;   // Vt row base for this (b,h)

  for (int s0 = 0; s0 < 2048; s0 += 32) {
    v8f sf0 = {}, sf1 = {};
    v16bf ka;
    ka = frag_from_bf16(Kb, b * 2048 + s0,      1024, colH,      lane);
    sf0 = wmma_bf16(ka, qf0, sf0);
    ka = frag_from_bf16(Kb, b * 2048 + s0,      1024, colH + 32, lane);
    sf0 = wmma_bf16(ka, qf1, sf0);
    ka = frag_from_bf16(Kb, b * 2048 + s0 + 16, 1024, colH,      lane);
    sf1 = wmma_bf16(ka, qf0, sf1);
    ka = frag_from_bf16(Kb, b * 2048 + s0 + 16, 1024, colH + 32, lane);
    sf1 = wmma_bf16(ka, qf1, sf1);

    float mx = m_run;
#pragma unroll
    for (int j = 0; j < 8; ++j) {
      sf0[j] *= scale; sf1[j] *= scale;
      mx = fmaxf(mx, fmaxf(sf0[j], sf1[j]));
    }
    mx = fmaxf(mx, __shfl_xor(mx, 16, 32));   // lanes L, L^16 share a query
    const float alpha = __expf(m_run - mx);
    m_run = mx;

    float p0[8], p1[8], ps = 0.0f;
#pragma unroll
    for (int j = 0; j < 8; ++j) {
      p0[j] = __expf(sf0[j] - mx);
      p1[j] = __expf(sf1[j] - mx);
      ps += p0[j] + p1[j];
    }
    ps += __shfl_xor(ps, 16, 32);
    l_run = alpha * l_run + ps;

    v16bf pf;                        // P^T B-fragment, built lane-locally
#pragma unroll
    for (int j = 0; j < 8; ++j) { pf[j] = (__bf16)p0[j]; pf[8 + j] = (__bf16)p1[j]; }
#pragma unroll
    for (int j = 0; j < 8; ++j) { o0[j] *= alpha; o1[j] *= alpha; o2[j] *= alpha; o3[j] *= alpha; }

    o0 = wmma_bf16(frag_from_bf16(Vt, vrow,      2048, s0, lane), pf, o0);
    o1 = wmma_bf16(frag_from_bf16(Vt, vrow + 16, 2048, s0, lane), pf, o1);
    o2 = wmma_bf16(frag_from_bf16(Vt, vrow + 32, 2048, s0, lane), pf, o2);
    o3 = wmma_bf16(frag_from_bf16(Vt, vrow + 48, 2048, s0, lane), pf, o3);
  }

  const float inv = 1.0f / l_run;
  const int q = q0 + (lane & 15);
  __bf16* dst = AO + (size_t)(b * 2048 + q) * 1024 + colH + ((lane >> 4) << 3);
  v8f oo[4] = {o0, o1, o2, o3};
#pragma unroll
  for (int t = 0; t < 4; ++t) {
    uint4 v;
    v.x = pack2_bf16(oo[t][0] * inv, oo[t][1] * inv);
    v.y = pack2_bf16(oo[t][2] * inv, oo[t][3] * inv);
    v.z = pack2_bf16(oo[t][4] * inv, oo[t][5] * inv);
    v.w = pack2_bf16(oo[t][6] * inv, oo[t][7] * inv);
    *reinterpret_cast<uint4*>(dst + t * 16) = v;
  }
}

// ---------------------------------------------------------------------------
// Output projection:  Y = AO(bf16) @ w4^T + b4 -> fp32 d_out  (32x64/wave)
// ---------------------------------------------------------------------------
__global__ __launch_bounds__(256) void out_proj_kernel(
    const __bf16* __restrict__ A, const __bf16* __restrict__ W,
    const float* __restrict__ bias, float* __restrict__ Out) {
  const int lane = threadIdx.x & 31;
  const int wave = blockIdx.x * (blockDim.x >> 5) + (threadIdx.x >> 5);
  const int tm = (wave >> 4) * 32;
  const int tn = (wave & 15) * 64;
  v8f c[2][4] = {};
  for (int k0 = 0; k0 < 1024; k0 += 32) {
    const v16bf a0 = frag_from_bf16(A, tm,      1024, k0, lane);
    const v16bf a1 = frag_from_bf16(A, tm + 16, 1024, k0, lane);
#pragma unroll
    for (int j = 0; j < 4; ++j) {
      const v16bf bj = frag_from_bf16(W, tn + j * 16, 1024, k0, lane);
      c[0][j] = wmma_bf16(a0, bj, c[0][j]);
      c[1][j] = wmma_bf16(a1, bj, c[1][j]);
    }
  }
#pragma unroll
  for (int i = 0; i < 2; ++i)
#pragma unroll
    for (int j = 0; j < 4; ++j) {
      const int n  = tn + j * 16 + (lane & 15);
      const float bn = bias[n];
      const int m0 = tm + i * 16 + ((lane >> 4) << 3);
#pragma unroll
      for (int r = 0; r < 8; ++r)
        Out[(size_t)(m0 + r) * 1024 + n] = c[i][j][r] + bn;
    }
}

// ---------------------------------------------------------------------------
extern "C" void kernel_launch(void* const* d_in, const int* in_sizes, int n_in,
                              void* d_out, int out_size, void* d_ws, size_t ws_size,
                              hipStream_t stream) {
  const float* query  = (const float*)d_in[0];
  const float* key_in = (const float*)d_in[1];
  const float* value  = (const float*)d_in[2];
  const float* w1 = (const float*)d_in[3];
  const float* w2 = (const float*)d_in[4];
  const float* w3 = (const float*)d_in[5];
  const float* w4 = (const float*)d_in[6];
  const float* b1 = (const float*)d_in[7];
  const float* b2 = (const float*)d_in[8];
  const float* b3 = (const float*)d_in[9];
  const float* b4 = (const float*)d_in[10];

  const size_t ACT = (size_t)4096 * 1024;   // activation elements
  const size_t WEL = (size_t)1024 * 1024;   // weight elements
  __bf16* Xq  = (__bf16*)d_ws;      // reused as AO after projections
  __bf16* Xk  = Xq  + ACT;
  __bf16* Xv  = Xk  + ACT;
  __bf16* W1b = Xv  + ACT;
  __bf16* W2b = W1b + WEL;
  __bf16* W3b = W2b + WEL;
  __bf16* W4b = W3b + WEL;
  __bf16* Qb  = W4b + WEL;
  __bf16* Kb  = Qb  + ACT;
  __bf16* Vt  = Kb  + ACT;
  __bf16* AO  = Xq;                 // alias: Xq dead after Q projection

  dim3 blk(256);
  const int actBlocks = (int)(ACT / 8 / 256);   // 2048
  const int wBlocks   = (int)(WEL / 8 / 256);   // 512
  cvt_f32_bf16_kernel<<<actBlocks, blk, 0, stream>>>(query,  Xq,  (int)(ACT / 8));
  cvt_f32_bf16_kernel<<<actBlocks, blk, 0, stream>>>(key_in, Xk,  (int)(ACT / 8));
  cvt_f32_bf16_kernel<<<actBlocks, blk, 0, stream>>>(value,  Xv,  (int)(ACT / 8));
  cvt_f32_bf16_kernel<<<wBlocks,   blk, 0, stream>>>(w1,     W1b, (int)(WEL / 8));
  cvt_f32_bf16_kernel<<<wBlocks,   blk, 0, stream>>>(w2,     W2b, (int)(WEL / 8));
  cvt_f32_bf16_kernel<<<wBlocks,   blk, 0, stream>>>(w3,     W3b, (int)(WEL / 8));
  cvt_f32_bf16_kernel<<<wBlocks,   blk, 0, stream>>>(w4,     W4b, (int)(WEL / 8));

  // 128x16 = 2048 wave-tiles (32x64) per GEMM / 8 waves per block
  proj_qkv_kernel<<<256, blk, 0, stream>>>(Xq, W1b, b1, Qb, 0);
  proj_qkv_kernel<<<256, blk, 0, stream>>>(Xk, W2b, b2, Kb, 0);
  proj_qkv_kernel<<<256, blk, 0, stream>>>(Xv, W3b, b3, Vt, 1);
  // 2*16*128 = 4096 query-tile waves -> 512 blocks
  flash_attn_kernel<<<512, blk, 0, stream>>>(Qb, Kb, Vt, AO);
  out_proj_kernel<<<256, blk, 0, stream>>>(AO, W4b, b4, (float*)d_out);
}
